// GNN_5798205850015
// MI455X (gfx1250) — matrix operand order repaired and checked by reading, not compiled, not run on hardware
//
#include <hip/hip_runtime.h>

// ---------------------------------------------------------------------------
// GatedGCN (2 layers), N=27 nodes, C=512, B=64, on gfx1250 (CDNA5, wave32).
// GEMMs run on v_wmma_f32_16x16x32_bf16 (fp32 accumulate); tile staging uses
// GLOBAL_LOAD_ASYNC_TO_LDS_B128 (ASYNCcnt) when the builtin is available.
// ---------------------------------------------------------------------------

typedef __attribute__((ext_vector_type(16))) __bf16 bf16x16;
typedef __attribute__((ext_vector_type(8)))  __bf16 bf16x8;
typedef __attribute__((ext_vector_type(4)))  __bf16 bf16x4;
typedef __attribute__((ext_vector_type(8)))  float  f32x8;

union FragAB { bf16x16 v; bf16x8 h[2]; uint4 u[2]; };

#define NN   27
#define EE   729          // 27*27
#define CC   512
#define BB   64
#define MNODE 1728        // B*N
#define MEDGE 46656       // B*N*N
#define LDA  40           // padded LDS row (bf16 elems): 80B stride -> conflict free
#define EPSV 1e-5f

#if defined(__HIP_DEVICE_COMPILE__) && \
    __has_builtin(__builtin_amdgcn_global_load_async_to_lds_b128)
#define ASYNC_LDS 1
// builtin signature (from clang diagnostic): param0 = int4 AS1*, param1 = int4 AS3*
typedef int int4v __attribute__((vector_size(16)));
typedef __attribute__((address_space(1))) int4v* g4ptr;
typedef __attribute__((address_space(3))) int4v* l4ptr;
#else
#define ASYNC_LDS 0
#endif

__device__ __forceinline__ void wait_async0() {
#if ASYNC_LDS
#if __has_builtin(__builtin_amdgcn_s_wait_asynccnt)
    __builtin_amdgcn_s_wait_asynccnt(0);
#else
    asm volatile("s_wait_asynccnt 0" ::: "memory");
#endif
#endif
}

// ---------------------------------------------------------------------------
// fp32 -> bf16 conversion (vec4)
// ---------------------------------------------------------------------------
__global__ __launch_bounds__(256) void cvt_f32_bf16(const float* __restrict__ s,
                                                    __bf16* __restrict__ d, int n4) {
    int i = blockIdx.x * 256 + threadIdx.x;
    if (i >= n4) return;
    float4 v = ((const float4*)s)[i];
    bf16x4 o = { (__bf16)v.x, (__bf16)v.y, (__bf16)v.z, (__bf16)v.w };
    ((bf16x4*)d)[i] = o;
}

// ---------------------------------------------------------------------------
// Y[M x Ntot] = X[M x 512] @ W^T  with W row-major [n_out, k] (so B-cols are
// contiguous in k).  Block tile 64x256, 8 waves, wave tile 32x64 (2x4 WMMA
// accumulators).  grid.x = M/64, grid.y = Ntot/256; weight selected per
// n-block so 4 distinct weights can be fused into one launch (ldY = 2048).
// Optional fused epilogue (edge GEMM): += Vix[b,i,col] + Vjx[b,j,col], saving
// a full read+write pass over the 95.5MB edge activation.
// ---------------------------------------------------------------------------
__global__ __launch_bounds__(256, 2) void gemm_bf16_wmma(
    const __bf16* __restrict__ X,
    const __bf16* __restrict__ W0, const __bf16* __restrict__ W1,
    const __bf16* __restrict__ W2, const __bf16* __restrict__ W3,
    float* __restrict__ Y, int ldY,
    const float* __restrict__ Vadd, int fuse_msg) {
    __shared__ __bf16 lA[2][64 * LDA];
    __shared__ __bf16 lB[2][256 * LDA];

    const int tid  = threadIdx.x;
    const int lane = tid & 31;
    const int wave = tid >> 5;
    const int wm   = wave & 1;   // 2 sub-tiles in M
    const int wn   = wave >> 1;  // 4 sub-tiles in N
    const int m0   = blockIdx.x * 64;
    const int nb   = blockIdx.y;

    const int wsel = nb >> 1;
    const __bf16* W = (wsel == 0) ? W0 : (wsel == 1) ? W1 : (wsel == 2) ? W2 : W3;
    const int wrow0 = (nb & 1) * 256;   // column offset inside selected W

    // global->LDS staging assignments
    const int arow = tid >> 2, ach = tid & 3;   // A: 64 rows x 4 chunks of 8 bf16
    const __bf16* gA = X + (size_t)(m0 + arow) * CC + ach * 8;
    const __bf16* gB = W + (size_t)(wrow0 + tid) * CC;
    __bf16* const lAdst = &lA[0][arow * LDA + ach * 8];
    __bf16* const lBdst = &lB[0][tid * LDA];
    const int lbuf = 64 * LDA;          // lA buffer stride (elems)
    const int lbufB = 256 * LDA;

    f32x8 acc[2][4];
    const f32x8 zero = {0.f, 0.f, 0.f, 0.f, 0.f, 0.f, 0.f, 0.f};
#pragma unroll
    for (int mt = 0; mt < 2; ++mt)
#pragma unroll
        for (int nt = 0; nt < 4; ++nt) acc[mt][nt] = zero;

#if ASYNC_LDS
    // stage k-tile (global -> LDS, no VGPR round-trip; tracked by ASYNCcnt)
    auto stage = [&](int buf, int k0) {
        __builtin_amdgcn_global_load_async_to_lds_b128(
            (g4ptr)(gA + k0), (l4ptr)(lAdst + buf * lbuf), 0, 0);
#pragma unroll
        for (int ch = 0; ch < 4; ++ch)
            __builtin_amdgcn_global_load_async_to_lds_b128(
                (g4ptr)(gB + k0 + ch * 8),
                (l4ptr)(lBdst + buf * lbufB + ch * 8), 0, 0);
    };
    stage(0, 0);
    wait_async0();
    __syncthreads();
#else
    {   // register-staged prime
        uint4 a0 = *(const uint4*)gA;
        *(uint4*)lAdst = a0;
#pragma unroll
        for (int ch = 0; ch < 4; ++ch) {
            uint4 b0 = *(const uint4*)(gB + ch * 8);
            *(uint4*)(lBdst + ch * 8) = b0;
        }
    }
    __syncthreads();
#endif

    for (int kk = 0; kk < 16; ++kk) {               // K = 512 / 32
        const int cur = kk & 1, nxt = cur ^ 1;
        const bool pf = (kk < 15);
#if ASYNC_LDS
        if (pf) stage(nxt, (kk + 1) * 32);          // fire-and-forget
#else
        uint4 ra, rb[4];
        if (pf) {                                   // issue next-tile loads early
            const int k0 = (kk + 1) * 32;
            ra = *(const uint4*)(gA + k0);
#pragma unroll
            for (int ch = 0; ch < 4; ++ch)
                rb[ch] = *(const uint4*)(gB + k0 + ch * 8);
        }
#endif
        // assemble fragments per documented CDNA5 16-bit layouts
        FragAB af[2], bfr[4];
#pragma unroll
        for (int mt = 0; mt < 2; ++mt) {
            const int r = wm * 32 + mt * 16 + (lane & 15);
            const int o = r * LDA + (lane >> 4) * 8;          // K base 0 / 8
            af[mt].h[0] = *(const bf16x8*)&lA[cur][o];         // K +0..7
            af[mt].h[1] = *(const bf16x8*)&lA[cur][o + 16];    // K +16..23
        }
#pragma unroll
        for (int nt = 0; nt < 4; ++nt) {
            const int r = wn * 64 + nt * 16 + (lane & 15);
            const int o = r * LDA + (lane >> 4) * 8;           // K base 0 / 8
            bfr[nt].h[0] = *(const bf16x8*)&lB[cur][o];
            bfr[nt].h[1] = *(const bf16x8*)&lB[cur][o + 16];
        }
#pragma unroll
        for (int mt = 0; mt < 2; ++mt)
#pragma unroll
            for (int nt = 0; nt < 4; ++nt)
                acc[mt][nt] = __builtin_amdgcn_wmma_f32_16x16x32_bf16(
                    false, af[mt].v, false, bfr[nt].v, (short)0, acc[mt][nt],
                    false, false);
#if ASYNC_LDS
        if (pf) wait_async0();                      // LDS writes landed
#else
        if (pf) {                                   // publish next tile
            *(uint4*)(lAdst + nxt * lbuf) = ra;
#pragma unroll
            for (int ch = 0; ch < 4; ++ch)
                *(uint4*)(lBdst + nxt * lbufB + ch * 8) = rb[ch];
        }
#endif
        // single barrier: frees cur for overwrite, publishes nxt for all waves
        __syncthreads();
    }

    // C/D layout: vgpr v, lanes 0-15 -> (M=v, N=lane); lanes 16-31 -> (M=v+8)
    const int colbase = nb * 256 + wn * 64 + (lane & 15);
    const int rowbase = m0 + wm * 32 + 8 * (lane >> 4);
    if (fuse_msg) {
        // edge GEMM epilogue: Y = acc + Vix[b,i,col] + Vjx[b,j,col]
#pragma unroll
        for (int mt = 0; mt < 2; ++mt)
#pragma unroll
            for (int v = 0; v < 8; ++v) {
                const int row = rowbase + mt * 16 + v;   // = b*729 + i*27 + j
                const int b = row / EE;
                const int e = row - b * EE;
                const int i = e / NN;
                const int j = e - i * NN;
                const float* vi = Vadd + (size_t)(b * NN + i) * 2048;
                const float* vj = Vadd + (size_t)(b * NN + j) * 2048 + 512;
#pragma unroll
                for (int nt = 0; nt < 4; ++nt) {
                    const int col = colbase + nt * 16;
                    Y[(size_t)row * ldY + col] =
                        acc[mt][nt][v] + vi[col] + vj[col];
                }
            }
    } else {
#pragma unroll
        for (int mt = 0; mt < 2; ++mt)
#pragma unroll
            for (int nt = 0; nt < 4; ++nt)
#pragma unroll
                for (int v = 0; v < 8; ++v)
                    Y[(size_t)(rowbase + mt * 16 + v) * ldY + colbase + nt * 16] =
                        acc[mt][nt][v];
    }
}

// ---------------------------------------------------------------------------
// BatchNorm stats: per "channel" e (block), reduce over (B=64, C=512).
// src index: (b*nchan + e)*512 + c.   Writes mean[e], rstd[e].
// ---------------------------------------------------------------------------
__global__ __launch_bounds__(256) void bn_stats(const float* __restrict__ src,
                                                float* __restrict__ mean,
                                                float* __restrict__ rstd,
                                                int nchan) {
    const int e = blockIdx.x, t = threadIdx.x;
    float s = 0.f, q = 0.f;
    for (int b = 0; b < BB; ++b) {
        float2 v = *(const float2*)&src[((size_t)b * nchan + e) * CC + t * 2];
        s += v.x + v.y;
        q += v.x * v.x + v.y * v.y;
    }
    __shared__ float rs[256], rq[256];
    rs[t] = s; rq[t] = q;
    __syncthreads();
    for (int k = 128; k > 0; k >>= 1) {
        if (t < k) { rs[t] += rs[t + k]; rq[t] += rq[t + k]; }
        __syncthreads();
    }
    if (t == 0) {
        float m = rs[0] * (1.f / 32768.f);
        float v = rq[0] * (1.f / 32768.f) - m * m;
        mean[e] = m;
        rstd[e] = rsqrtf(v + EPSV);
    }
}

// ---------------------------------------------------------------------------
// out = res + relu(g*(src-mean)*rstd + b), per-channel e = (idx/512)%nchan.
// Optionally also emits bf16 copy (for next layer's GEMM input).  vec4.
// ---------------------------------------------------------------------------
__global__ __launch_bounds__(256) void bn_residual_relu(
    const float* __restrict__ res, const float* __restrict__ src,
    const float* __restrict__ mean, const float* __restrict__ rstd,
    const float* __restrict__ gam, const float* __restrict__ bet,
    float* __restrict__ out, __bf16* __restrict__ out_bf, int write_bf,
    int nchan, int total4) {
    int t = blockIdx.x * 256 + threadIdx.x;
    if (t >= total4) return;
    size_t idx = (size_t)t * 4;
    int e = (int)((idx / CC) % nchan);
    float m = mean[e], r = rstd[e], g = gam[e], bb = bet[e];
    float4 sv = ((const float4*)src)[t];
    float4 rv = ((const float4*)res)[t];
    float4 o;
    o.x = rv.x + fmaxf(g * (sv.x - m) * r + bb, 0.f);
    o.y = rv.y + fmaxf(g * (sv.y - m) * r + bb, 0.f);
    o.z = rv.z + fmaxf(g * (sv.z - m) * r + bb, 0.f);
    o.w = rv.w + fmaxf(g * (sv.w - m) * r + bb, 0.f);
    ((float4*)out)[t] = o;
    if (write_bf) {
        bf16x4 ob = { (__bf16)o.x, (__bf16)o.y, (__bf16)o.z, (__bf16)o.w };
        ((bf16x4*)out_bf)[t] = ob;
    }
}

// ---------------------------------------------------------------------------
// Attention + aggregation, fully fused per (b,i,c):
//   w_j = exp(sigmoid(edge[b,i,j,c]));  agg = sum_j w_j*Ujx[b,j,c] / sum_j w_j
//   xpre = Uix[b,i,c] + agg/27
// ---------------------------------------------------------------------------
__global__ __launch_bounds__(256) void attn_agg(const float* __restrict__ edge,
                                                const float* __restrict__ Vout,
                                                float* __restrict__ xpre) {
    int t = blockIdx.x * 256 + threadIdx.x;      // 0 .. B*N*C-1
    int c  = t % CC;
    int bi = t / CC;                              // b*27 + i
    int b  = bi / NN, i = bi % NN;
    const float* ebase = edge + ((size_t)b * EE + (size_t)i * NN) * CC + c;
    const float* ubase = Vout + (size_t)(b * NN) * 2048 + 1536 + c;   // Ujx (WV)
    float den = 0.f, acc = 0.f;
#pragma unroll
    for (int j = 0; j < NN; ++j) {
        float s  = ebase[(size_t)j * CC];
        float sg = 1.f / (1.f + __expf(-s));
        float w  = __expf(sg);
        den += w;
        acc += w * ubase[(size_t)j * 2048];
    }
    float uix = Vout[(size_t)bi * 2048 + 1024 + c];                    // Uix (WU)
    xpre[t] = uix + (acc / den) * (1.f / (float)NN);
}

// ---------------------------------------------------------------------------
extern "C" void kernel_launch(void* const* d_in, const int* in_sizes, int n_in,
                              void* d_out, int out_size, void* d_ws, size_t ws_size,
                              hipStream_t stream) {
    (void)in_sizes; (void)n_in; (void)out_size; (void)ws_size;
    const float* x_in    = (const float*)d_in[0];
    const float* edge_in = (const float*)d_in[1];

    // workspace carve-out (256B aligned)
    char*  ws  = (char*)d_ws;
    size_t off = 0;
    auto carve = [&](size_t bytes) -> char* {
        char* p = ws + off;
        off += (bytes + 255) & ~(size_t)255;
        return p;
    };
    __bf16* Wb   = (__bf16*)carve((size_t)10 * CC * CC * 2);        // 10 weights, bf16
    __bf16* xb   = (__bf16*)carve((size_t)MNODE * CC * 2);
    __bf16* eb   = (__bf16*)carve((size_t)MEDGE * CC * 2);
    float*  Vout = (float*)carve((size_t)MNODE * 2048 * 4);
    float*  elin = (float*)carve((size_t)MEDGE * CC * 4);
    float*  xpre = (float*)carve((size_t)MNODE * CC * 4);
    float*  meanE = (float*)carve(EE * 4);
    float*  rstdE = (float*)carve(EE * 4);
    float*  meanV = (float*)carve(NN * 4);
    float*  rstdV = (float*)carve(NN * 4);

    float* x_out = (float*)d_out;                       // [1728*512]
    float* e_out = (float*)d_out + (size_t)MNODE * CC;  // [46656*512]

    // convert weights (order per layer: WA, WB, WU, WV, WE)
    const int widx[10] = {4, 5, 2, 3, 6, 9, 10, 7, 8, 11};
    for (int k = 0; k < 10; ++k)
        cvt_f32_bf16<<<256, 256, 0, stream>>>((const float*)d_in[widx[k]],
                                              Wb + (size_t)k * CC * CC, 65536);
    cvt_f32_bf16<<<864, 256, 0, stream>>>(x_in, xb, MNODE * CC / 4);
    cvt_f32_bf16<<<23328, 256, 0, stream>>>(edge_in, eb, MEDGE * CC / 4);

    const int e_total4 = MEDGE * CC / 4;   // 5971968
    const int n_total4 = MNODE * CC / 4;   // 221184

    for (int L = 0; L < 2; ++L) {
        const __bf16* WA = Wb + (size_t)(L * 5 + 0) * CC * CC;
        const __bf16* WBp = Wb + (size_t)(L * 5 + 1) * CC * CC;
        const __bf16* WU = Wb + (size_t)(L * 5 + 2) * CC * CC;
        const __bf16* WV = Wb + (size_t)(L * 5 + 3) * CC * CC;
        const __bf16* WE = Wb + (size_t)(L * 5 + 4) * CC * CC;
        const float* gv = (const float*)d_in[12 + L * 4];
        const float* bv = (const float*)d_in[13 + L * 4];
        const float* ge = (const float*)d_in[14 + L * 4];
        const float* be = (const float*)d_in[15 + L * 4];
        const float* xres = (L == 0) ? x_in : x_out;       // residual inputs
        const float* eres = (L == 0) ? edge_in : e_out;    // (L2: in-place)
        const int wr_bf = (L == 0) ? 1 : 0;                // L2 output stays fp32

        // fused node GEMM: Vout = x @ [WA^T | WB^T | WU^T | WV^T]
        gemm_bf16_wmma<<<dim3(MNODE / 64, 8), 256, 0, stream>>>(
            xb, WA, WBp, WU, WV, Vout, 2048, nullptr, 0);
        // edge GEMM with fused msg epilogue:
        //   elin = edge @ WE^T + Vix[b,i,:] + Vjx[b,j,:]
        gemm_bf16_wmma<<<dim3(MEDGE / 64, 2), 256, 0, stream>>>(
            eb, WE, WE, WE, WE, elin, 512, Vout, 1);
        // edge BN (per edge-slot e over B,C) + residual + relu -> e_out (+bf16)
        bn_stats<<<EE, 256, 0, stream>>>(elin, meanE, rstdE, EE);
        bn_residual_relu<<<23328, 256, 0, stream>>>(eres, elin, meanE, rstdE,
                                                    ge, be, e_out, eb, wr_bf,
                                                    EE, e_total4);
        // attention softmax(j) over sigmoid(e_out), aggregate, + Uix
        attn_agg<<<3456, 256, 0, stream>>>(e_out, Vout, xpre);
        // node BN (per node i over B,C) + residual + relu -> x_out (+bf16)
        bn_stats<<<NN, 256, 0, stream>>>(xpre, meanV, rstdV, NN);
        bn_residual_relu<<<864, 256, 0, stream>>>(xres, xpre, meanV, rstdV,
                                                  gv, bv, x_out, xb, wr_bf,
                                                  NN, n_total4);
    }
}
